// MultiHeadDotProductAttention_13408887898113
// MI455X (gfx1250) — matrix operand order repaired and checked by reading, not compiled, not run on hardware
//
#include <hip/hip_runtime.h>
#include <hip/hip_bf16.h>

// ---------------------------------------------------------------------------
// MHA on gfx1250 via bf16 WMMA (v_wmma_f32_16x16x32_bf16), f32 accumulate.
// B=2, S=2048, D=1024, H=16, hd=64. ~69 GFLOP vs ~50MB traffic -> compute
// bound -> all four GEMM-like stages on the 16x16x32 bf16 matrix pipe.
// Staging uses the CDNA5 async global->LDS path (global_load_async_to_lds_b128
// + s_wait_asynccnt) with a manually unrolled 2-deep pipeline so the buffer
// index is a literal: no per-iteration accumulator re-homing, constant LDS
// offsets, HBM/L2 latency hidden under the WMMAs.
// ---------------------------------------------------------------------------

typedef __bf16 bf16_t;
typedef __attribute__((ext_vector_type(8)))  __bf16 bf16x8;
typedef __attribute__((ext_vector_type(16))) __bf16 bf16x16;
typedef __attribute__((ext_vector_type(8)))  float  f32x8;

#define BATCH   2
#define SEQ     2048
#define DMODEL  1024
#define NHEADS  16
#define HDIM    64
#define MTOT    (BATCH * SEQ)

// Async global->LDS copy, 16B per lane. VDST operand is the per-lane LDS byte
// offset (LDS_BASE-relative == low 32 bits of a shared-space pointer).
__device__ __forceinline__ void async_copy_b128(void* lds_dst, const void* gsrc) {
    unsigned ldsoff = (unsigned)(size_t)lds_dst;
    asm volatile("global_load_async_to_lds_b128 %0, %1, off"
                 :: "v"(ldsoff), "v"(gsrc) : "memory");
}
// Wait until at most N async instructions are still outstanding (in-order).
#define WAIT_ASYNC(n) asm volatile("s_wait_asynccnt " #n ::: "memory")

__device__ __forceinline__ bf16x16 cat16(bf16x8 lo, bf16x8 hi) {
    return __builtin_shufflevector(lo, hi, 0,1,2,3,4,5,6,7,8,9,10,11,12,13,14,15);
}

// A-matrix 16x32 bf16 fragment (row-major tile, ld elems/row). ISA layout:
// lane holds row m=lane%16; K = (lane/16)*8+{0..7} and +16. Two 16B loads.
__device__ __forceinline__ bf16x16 load_a_frag(const bf16_t* base, int ld, int lane) {
    int m  = lane & 15;
    int hs = (lane >> 4) & 1;
    const bf16_t* row = base + m * ld + hs * 8;
    bf16x8 lo = *(const bf16x8*)(row);
    bf16x8 hi = *(const bf16x8*)(row + 16);
    return cat16(lo, hi);
}

// B-matrix 32x16 fragment from a transposed [N][K] tile: lane holds column
// n=lane%16; K = (lane/16)*8+{0..7} and +16. Two 16B loads.
__device__ __forceinline__ bf16x16 load_b_frag(const bf16_t* base, int ld, int lane) {
    int n  = lane & 15;
    int ks = (lane >> 4) & 1;
    const bf16_t* row = base + n * ld + ks * 8;
    bf16x8 lo = *(const bf16x8*)(row);
    bf16x8 hi = *(const bf16x8*)(row + 16);
    return cat16(lo, hi);
}

// ------------------------------- prep kernels ------------------------------

__global__ void cast_f32_to_bf16(const float* __restrict__ in,
                                 bf16_t* __restrict__ out, int n) {
    for (int i = blockIdx.x * blockDim.x + threadIdx.x; i < n;
         i += gridDim.x * blockDim.x)
        out[i] = (bf16_t)in[i];
}

// Wt[n][k] = W[k][n], cast to bf16.
__global__ void transpose_cast(const float* __restrict__ W,
                               bf16_t* __restrict__ Wt, int K, int N) {
    int total = K * N;
    for (int i = blockIdx.x * blockDim.x + threadIdx.x; i < total;
         i += gridDim.x * blockDim.x) {
        int n = i / K, k = i % K;
        Wt[i] = (bf16_t)W[(size_t)k * N + n];
    }
}

// ------------------------------- GEMM kernel -------------------------------
// C[M=4096][N=1024] = X[M][1024] * Wt[N][1024]^T (+bias, scatter per mode).
// Block tile 128x128, 8 waves of 32x64 (two A-frags share each B-frag),
// K-chunks of 32, double-buffered async LDS staging (4 asyncs/thread/chunk).
// mode 0: Q -> (acc+b)*0.125 -> Qs[b][h][s][f]
// mode 1: K -> Kt[b][h][s][f]
// mode 2: V -> Vt[b][h][f][s]
// mode 3: out-projection -> f32 out[m][n] = acc + bo[n]
__global__ void wmma_gemm(const bf16_t* __restrict__ X,
                          const bf16_t* __restrict__ Wt,
                          const float* __restrict__ bias,
                          bf16_t* __restrict__ outb,
                          float* __restrict__ outf,
                          int mode) {
    __shared__ bf16_t As[2][128][40];    // +8 pad -> conflict-free row starts
    __shared__ bf16_t Bs[2][128][40];

    int tid  = threadIdx.x;
    int lane = tid & 31;
    int wave = tid >> 5;
    int wm = wave & 3;                   // 4 waves along M (32 rows each)
    int wn = wave >> 2;                  // 2 waves along N (64 cols each)
    int m0 = blockIdx.x * 128;
    int n0 = blockIdx.y * 128;

    // staging: A and B are each 128 rows x 4 segs = 512 16B units
    int row0 = tid >> 2, row1 = row0 + 64, seg = tid & 3;

    f32x8 acc[2][4] = {};

    auto stage = [&](int buf, int k0) {  // 4 async instructions / thread
        async_copy_b128(&As[buf][row0][seg * 8],
                        X + (size_t)(m0 + row0) * DMODEL + k0 + seg * 8);
        async_copy_b128(&As[buf][row1][seg * 8],
                        X + (size_t)(m0 + row1) * DMODEL + k0 + seg * 8);
        async_copy_b128(&Bs[buf][row0][seg * 8],
                        Wt + (size_t)(n0 + row0) * DMODEL + k0 + seg * 8);
        async_copy_b128(&Bs[buf][row1][seg * 8],
                        Wt + (size_t)(n0 + row1) * DMODEL + k0 + seg * 8);
    };

    auto compute = [&](int buf) {        // 8 WMMAs, 12 ds_load_b128
        bf16x16 a0 = load_a_frag(&As[buf][wm * 32][0], 40, lane);
        bf16x16 a1 = load_a_frag(&As[buf][wm * 32 + 16][0], 40, lane);
        #pragma unroll
        for (int nt = 0; nt < 4; ++nt) {
            bf16x16 b = load_b_frag(&Bs[buf][wn * 64 + nt * 16][0], 40, lane);
            acc[0][nt] = __builtin_amdgcn_wmma_f32_16x16x32_bf16(
                false, a0, false, b, (short)0, acc[0][nt], false, false);
            acc[1][nt] = __builtin_amdgcn_wmma_f32_16x16x32_bf16(
                false, a1, false, b, (short)0, acc[1][nt], false, false);
        }
    };

    const int NCHUNK = DMODEL / 32;      // 32 (even)
    stage(0, 0);
    for (int it = 0; it < NCHUNK; it += 2) {
        // even half: compute buf 0, prefetch buf 1
        stage(1, (it + 1) * 32);
        WAIT_ASYNC(4);                   // buf0's group has landed
        __syncthreads();
        compute(0);
        __syncthreads();                 // buf0 readers done before refill
        // odd half: compute buf 1, prefetch buf 0
        if (it + 2 < NCHUNK) { stage(0, (it + 2) * 32); WAIT_ASYNC(4); }
        else                 { WAIT_ASYNC(0); }
        __syncthreads();
        compute(1);
        __syncthreads();
    }

    // Epilogue: C-frag element r -> (m = r + 8*(lane/16), n = lane%16)
    int hs = (lane >> 4) & 1;
    #pragma unroll
    for (int am = 0; am < 2; ++am)
        #pragma unroll
        for (int nt = 0; nt < 4; ++nt)
            #pragma unroll
            for (int r = 0; r < 8; ++r) {
                int gm = m0 + wm * 32 + am * 16 + r + 8 * hs;
                int gn = n0 + wn * 64 + nt * 16 + (lane & 15);
                float v = acc[am][nt][r] + bias[gn];
                if (mode == 3) {
                    outf[(size_t)gm * DMODEL + gn] = v;
                } else {
                    int b = gm >> 11, s = gm & (SEQ - 1);
                    int h = gn >> 6,  f = gn & (HDIM - 1);
                    size_t bh = (size_t)(b * NHEADS + h);
                    if (mode == 0) {
                        outb[(bh * SEQ + s) * HDIM + f] = (bf16_t)(v * 0.125f);
                    } else if (mode == 1) {
                        outb[(bh * SEQ + s) * HDIM + f] = (bf16_t)v;
                    } else {
                        outb[(bh * HDIM + f) * SEQ + s] = (bf16_t)v;  // V^T
                    }
                }
            }
}

// ----------------------------- attention kernel ----------------------------
// grid (S/128, H, B), 256 threads. Each wave: one 16-row Q tile, streaming
// 64-key chunks (double-buffered async K/V staging, literal buffer index);
// scores + online softmax + PV, all via bf16 WMMA.
__global__ void flash_attn(const bf16_t* __restrict__ Q,
                           const bf16_t* __restrict__ Kt,
                           const bf16_t* __restrict__ Vt,
                           bf16_t* __restrict__ Ctx) {
    __shared__ bf16_t Ks[2][64][72];     // keys  [sk][hd]
    __shared__ bf16_t Vs[2][64][72];     // V^T   [hd][sk]
    __shared__ bf16_t Ps[8][16][72];     // per-wave P staging [m][sk_local]

    int tid  = threadIdx.x;
    int lane = tid & 31;
    int wave = tid >> 5;
    int b = blockIdx.z, h = blockIdx.y;
    int q0 = blockIdx.x * 128 + wave * 16;

    size_t bh = (size_t)(b * NHEADS + h);
    const bf16_t* Qbh = Q  + bh * SEQ * HDIM;
    const bf16_t* Kbh = Kt + bh * SEQ * HDIM;
    const bf16_t* Vbh = Vt + bh * HDIM * SEQ;

    // staging: K and V tiles are 64 rows x 8 segs = 512 16B units each
    int r0 = tid >> 3, r1 = r0 + 32, seg = tid & 7;

    auto stage = [&](int buf, int sk0) { // 4 async instructions / thread
        async_copy_b128(&Ks[buf][r0][seg * 8],
                        Kbh + (size_t)(sk0 + r0) * HDIM + seg * 8);
        async_copy_b128(&Ks[buf][r1][seg * 8],
                        Kbh + (size_t)(sk0 + r1) * HDIM + seg * 8);
        async_copy_b128(&Vs[buf][r0][seg * 8],
                        Vbh + (size_t)r0 * SEQ + sk0 + seg * 8);
        async_copy_b128(&Vs[buf][r1][seg * 8],
                        Vbh + (size_t)r1 * SEQ + sk0 + seg * 8);
    };

    // Preload the wave's Q A-fragments (hd=64 -> 2 K-steps of 32).
    int m  = lane & 15;
    int hs = (lane >> 4) & 1;
    bf16x16 qa[2];
    #pragma unroll
    for (int kk = 0; kk < 2; ++kk) {
        const bf16_t* row = Qbh + (size_t)(q0 + m) * HDIM + kk * 32 + hs * 8;
        qa[kk] = cat16(*(const bf16x8*)(row), *(const bf16x8*)(row + 16));
    }

    float rmax[8], rsum[8];
    f32x8 acc[4] = {};
    #pragma unroll
    for (int r = 0; r < 8; ++r) { rmax[r] = -3.0e38f; rsum[r] = 0.0f; }

    auto process = [&](int buf) {
        // scores: 16x64 tile = 4 N-tiles x 2 K-steps
        f32x8 sc[4] = {};
        #pragma unroll
        for (int nt = 0; nt < 4; ++nt)
            #pragma unroll
            for (int kk = 0; kk < 2; ++kk) {
                bf16x16 bfrag = load_b_frag(&Ks[buf][nt * 16][kk * 32], 72, lane);
                sc[nt] = __builtin_amdgcn_wmma_f32_16x16x32_bf16(
                    false, qa[kk], false, bfrag, (short)0, sc[nt], false, false);
            }

        // online softmax: row stats live across the 16 lanes of a half-wave
        float alpha[8];
        #pragma unroll
        for (int r = 0; r < 8; ++r) {
            float tm = sc[0][r];
            #pragma unroll
            for (int nt = 1; nt < 4; ++nt) tm = fmaxf(tm, sc[nt][r]);
            #pragma unroll
            for (int w = 1; w < 16; w <<= 1)
                tm = fmaxf(tm, __shfl_xor(tm, w, 32));
            float nm = fmaxf(rmax[r], tm);
            alpha[r] = __expf(rmax[r] - nm);
            rmax[r]  = nm;
        }
        #pragma unroll
        for (int r = 0; r < 8; ++r) {
            float ps = 0.0f;
            int prow = r + 8 * hs;
            #pragma unroll
            for (int nt = 0; nt < 4; ++nt) {
                float p = __expf(sc[nt][r] - rmax[r]);
                ps += p;
                Ps[wave][prow][nt * 16 + (lane & 15)] = (bf16_t)p;
            }
            #pragma unroll
            for (int w = 1; w < 16; w <<= 1)
                ps += __shfl_xor(ps, w, 32);
            rsum[r] = alpha[r] * rsum[r] + ps;
            #pragma unroll
            for (int nt = 0; nt < 4; ++nt) acc[nt][r] *= alpha[r];
        }
        __syncthreads();   // make P visible for cross-lane A-frag reads

        // PV: ctx(16xhd) += P(16x64) * V(64xhd);  B^T = Vs[hd][sk]
        #pragma unroll
        for (int kk = 0; kk < 2; ++kk) {
            bf16x16 pa = load_a_frag(&Ps[wave][0][kk * 32], 72, lane);
            #pragma unroll
            for (int nt = 0; nt < 4; ++nt) {
                bf16x16 bfrag = load_b_frag(&Vs[buf][nt * 16][kk * 32], 72, lane);
                acc[nt] = __builtin_amdgcn_wmma_f32_16x16x32_bf16(
                    false, pa, false, bfrag, (short)0, acc[nt], false, false);
            }
        }
    };

    const int NIT = SEQ / 64;            // 32 (even)
    stage(0, 0);
    for (int it = 0; it < NIT; it += 2) {
        stage(1, (it + 1) * 64);
        WAIT_ASYNC(4);
        __syncthreads();
        process(0);
        __syncthreads();                 // buf0/P readers done before refill
        if (it + 2 < NIT) { stage(0, (it + 2) * 64); WAIT_ASYNC(4); }
        else              { WAIT_ASYNC(0); }
        __syncthreads();
        process(1);
        __syncthreads();
    }

    // finalize: divide by row sums, store ctx as [B][S][H*hd] bf16
    #pragma unroll
    for (int nt = 0; nt < 4; ++nt)
        #pragma unroll
        for (int r = 0; r < 8; ++r) {
            int s = q0 + r + 8 * hs;
            float v = acc[nt][r] / rsum[r];
            Ctx[((size_t)(b * SEQ + s)) * DMODEL + h * HDIM + nt * 16 + (lane & 15)] =
                (bf16_t)v;
        }
}

// ------------------------------- launcher ----------------------------------

extern "C" void kernel_launch(void* const* d_in, const int* in_sizes, int n_in,
                              void* d_out, int out_size, void* d_ws, size_t ws_size,
                              hipStream_t stream) {
    const float* inq  = (const float*)d_in[0];
    const float* inkv = (const float*)d_in[1];
    const float* Wq   = (const float*)d_in[2];
    const float* Wk   = (const float*)d_in[3];
    const float* Wv   = (const float*)d_in[4];
    const float* bq   = (const float*)d_in[5];
    const float* bk   = (const float*)d_in[6];
    const float* bv   = (const float*)d_in[7];
    const float* Wo   = (const float*)d_in[8];
    const float* bo   = (const float*)d_in[9];
    float* out = (float*)d_out;

    const size_t NX = (size_t)MTOT * DMODEL;      // 4M elems
    const size_t NW = (size_t)DMODEL * DMODEL;    // 1M elems
    bf16_t* ws   = (bf16_t*)d_ws;
    bf16_t* Xq   = ws;
    bf16_t* Xkv  = Xq  + NX;
    bf16_t* WqT  = Xkv + NX;
    bf16_t* WkT  = WqT + NW;
    bf16_t* WvT  = WkT + NW;
    bf16_t* WoT  = WvT + NW;
    bf16_t* Qs   = WoT + NW;
    bf16_t* Kts  = Qs  + NX;
    bf16_t* Vts  = Kts + NX;
    bf16_t* Ctx  = Vts + NX;

    cast_f32_to_bf16<<<4096, 256, 0, stream>>>(inq,  Xq,  (int)NX);
    cast_f32_to_bf16<<<4096, 256, 0, stream>>>(inkv, Xkv, (int)NX);
    transpose_cast<<<4096, 256, 0, stream>>>(Wq, WqT, DMODEL, DMODEL);
    transpose_cast<<<4096, 256, 0, stream>>>(Wk, WkT, DMODEL, DMODEL);
    transpose_cast<<<4096, 256, 0, stream>>>(Wv, WvT, DMODEL, DMODEL);
    transpose_cast<<<4096, 256, 0, stream>>>(Wo, WoT, DMODEL, DMODEL);

    dim3 ggrid(MTOT / 128, DMODEL / 128);
    wmma_gemm<<<ggrid, 256, 0, stream>>>(Xq,  WqT, bq, Qs,  nullptr, 0);
    wmma_gemm<<<ggrid, 256, 0, stream>>>(Xkv, WkT, bk, Kts, nullptr, 1);
    wmma_gemm<<<ggrid, 256, 0, stream>>>(Xkv, WvT, bv, Vts, nullptr, 2);

    flash_attn<<<dim3(SEQ / 128, NHEADS, BATCH), 256, 0, stream>>>(Qs, Kts, Vts, Ctx);

    wmma_gemm<<<ggrid, 256, 0, stream>>>(Ctx, WoT, bo, nullptr, out, 3);
}
